// SelfAttention_45715631899272
// MI455X (gfx1250) — matrix operand order repaired and checked by reading, not compiled, not run on hardware
//
#include <hip/hip_runtime.h>
#include <hip/hip_bf16.h>
#include <math.h>

// ---------------------------------------------------------------------------
// Problem constants (match reference)
// ---------------------------------------------------------------------------
#define DIMN      1024
#define NHEADS    16
#define HEADD     64
#define BATCH     2
#define SEQ       2048
#define MROWS     (BATCH * SEQ)        // 4096
#define QKVN      (3 * DIMN)           // 3072
#define INV_SCALE 0.125f               // 1/sqrt(64)
#define LN_EPS    1e-5f

typedef __attribute__((ext_vector_type(16))) __bf16 v16bf;
typedef __attribute__((ext_vector_type(8)))  __bf16 v8bf;
typedef __attribute__((ext_vector_type(8)))  float  v8f;

// ---------------------------------------------------------------------------
// WMMA helper (CDNA5: V_WMMA_F32_16X16X32_BF16, wave32)
// ---------------------------------------------------------------------------
__device__ __forceinline__ v8f wmma_bf16(v16bf a, v16bf b, v8f c) {
    return __builtin_amdgcn_wmma_f32_16x16x32_bf16(
        false, a, false, b, (short)0, c, false, false);
}

__device__ __forceinline__ v16bf cat8(v8bf lo, v8bf hi) {
    return __builtin_shufflevector(lo, hi, 0,1,2,3,4,5,6,7,8,9,10,11,12,13,14,15);
}

// A fragment: 16x32 bf16 tile, row-major LDS (ld % 8 == 0).
// lanes 0-15 -> M=lane, K={0..7,16..23}; lanes 16-31 -> M=lane-16, K={8..15,24..31}
__device__ __forceinline__ v16bf load_a_frag(const __bf16* base, int ld,
                                             int row0, int k0) {
    const int lane = threadIdx.x & 31;
    const __bf16* p = base + (row0 + (lane & 15)) * ld + k0 + ((lane >> 4) << 3);
    return cat8(*(const v8bf*)(p), *(const v8bf*)(p + 16));
}

// B fragment: 32x16 (K x N) from a TRANSPOSED LDS tile Bt[col][k] (ld % 8 == 0).
// lanes 0-15 -> N=lane, K={0..7,16..23}; lanes 16-31 -> N=lane-16, K={8..15,24..31}
__device__ __forceinline__ v16bf load_bt_frag(const __bf16* baseT, int ld,
                                              int n0, int k0) {
    const int lane = threadIdx.x & 31;
    const __bf16* p = baseT + (n0 + (lane & 15)) * ld + k0 + ((lane >> 4) << 3);
    return cat8(*(const v8bf*)(p), *(const v8bf*)(p + 16));
}

__device__ __forceinline__ float elu1(float x) {
    return x > 0.f ? x : (expf(x) - 1.f);
}

// ---------------------------------------------------------------------------
// CDNA5 async memory->LDS copy (ASYNCcnt tracked, bypasses VGPRs).
// Per-lane: LDS[lds_off] = MEM[gaddr], 16 bytes.
// ---------------------------------------------------------------------------
__device__ __forceinline__ void async_copy_b128(const void* g, const void* lds) {
    const unsigned int      loff  = (unsigned int)(unsigned long long)(uintptr_t)lds;
    const unsigned long long gadr = (unsigned long long)(uintptr_t)g;
    asm volatile("global_load_async_to_lds_b128 %0, %1, off"
                 :: "v"(loff), "v"(gadr) : "memory");
}
__device__ __forceinline__ void wait_async0() {
    asm volatile("s_wait_asynccnt 0" ::: "memory");
}

// ---------------------------------------------------------------------------
// GEMM tile staging (f32 -> bf16), register-batched so global loads issue
// back-to-back; conversion + LDS store deferred (software pipelining).
// A tile: 64x32 (ldA = DIMN).  B tile: 32x64 from w (ldw), stored transposed.
// ---------------------------------------------------------------------------
__device__ __forceinline__ void gemm_stage_load(const float* __restrict__ A,
        const float* __restrict__ W, int ldw, int m0, int n0, int k0, int tid,
        float4* ra, float4* rb) {
#pragma unroll
    for (int i = 0; i < 4; ++i) {
        int q = tid + i * 128;
        ra[i] = *(const float4*)(A + (size_t)(m0 + (q >> 3)) * DIMN + k0 + (q & 7) * 4);
    }
#pragma unroll
    for (int i = 0; i < 4; ++i) {
        int q = tid + i * 128;
        rb[i] = *(const float4*)(W + (size_t)(k0 + (q >> 4)) * ldw + n0 + (q & 15) * 4);
    }
}

__device__ __forceinline__ void gemm_stage_store(__bf16* As, __bf16* Bts, int tid,
        const float4* ra, const float4* rb) {
#pragma unroll
    for (int i = 0; i < 4; ++i) {
        int q = tid + i * 128;
        __bf16* d = As + (q >> 3) * 40 + (q & 7) * 4;
        d[0] = (__bf16)ra[i].x; d[1] = (__bf16)ra[i].y;
        d[2] = (__bf16)ra[i].z; d[3] = (__bf16)ra[i].w;
    }
#pragma unroll
    for (int i = 0; i < 4; ++i) {
        int q = tid + i * 128;
        int r = q >> 4, c = (q & 15) * 4;
        Bts[(c + 0) * 40 + r] = (__bf16)rb[i].x;
        Bts[(c + 1) * 40 + r] = (__bf16)rb[i].y;
        Bts[(c + 2) * 40 + r] = (__bf16)rb[i].z;
        Bts[(c + 3) * 40 + r] = (__bf16)rb[i].w;
    }
}

// ---------------------------------------------------------------------------
// Kernel 1: QKV projection.  C = x @ w_qkv + b_qkv  (4096 x 3072, K=1024).
// Double-buffered LDS, 64x64 tile / WG (4 waves), stride-3 de-interleave
// epilogue into head-major bf16 Q (pre-scaled), K, V (ELU).
// ---------------------------------------------------------------------------
__global__ __launch_bounds__(128) void qkv_gemm_kernel(
    const float* __restrict__ x, const float* __restrict__ w,
    const float* __restrict__ bias,
    __bf16* __restrict__ Qo, __bf16* __restrict__ Ko, __bf16* __restrict__ Vo)
{
    __shared__ alignas(16) __bf16 As [2][64 * 40];
    __shared__ alignas(16) __bf16 Bts[2][64 * 40];

    const int tid  = threadIdx.x;
    const int lane = tid & 31;
    const int wave = tid >> 5;
    const int m0   = blockIdx.y * 64;
    const int n0   = blockIdx.x * 64;

    v8f acc[4] = {};
    float4 ra[4], rb[4];

    gemm_stage_load(x, w, QKVN, m0, n0, 0, tid, ra, rb);
    gemm_stage_store(As[0], Bts[0], tid, ra, rb);
    __syncthreads();

    int cur = 0;
    for (int k0 = 0; k0 < DIMN; k0 += 32) {
        const bool has_next = (k0 + 32 < DIMN);
        if (has_next) {
            gemm_stage_load(x, w, QKVN, m0, n0, k0 + 32, tid, ra, rb);
            __builtin_prefetch(w + (size_t)(k0 + 64) * QKVN + n0 + (lane << 2), 0, 1);
        }
        v16bf a = load_a_frag(As[cur], 40, wave * 16, 0);
#pragma unroll
        for (int nb = 0; nb < 4; ++nb)
            acc[nb] = wmma_bf16(a, load_bt_frag(Bts[cur], 40, nb * 16, 0), acc[nb]);
        if (has_next) {
            gemm_stage_store(As[cur ^ 1], Bts[cur ^ 1], tid, ra, rb);
            cur ^= 1;
        }
        __syncthreads();
    }

    // epilogue: bias + stride-3 split (n%3 -> Q/K/V), head split
    const int half = (lane >> 4) << 3;
    const int ncol = lane & 15;
#pragma unroll
    for (int nb = 0; nb < 4; ++nb) {
        const int n  = n0 + nb * 16 + ncol;
        const float bv = bias[n];
        const int c  = n % 3;
        const int d  = n / 3;
        const int h  = d >> 6;
        const int hd = d & 63;
#pragma unroll
        for (int r = 0; r < 8; ++r) {
            const int grow = m0 + wave * 16 + r + half;
            const int b    = grow >> 11;
            const int s    = grow & (SEQ - 1);
            const size_t idx = (((size_t)(b * NHEADS + h) * SEQ) + s) * HEADD + hd;
            const float val = acc[nb][r] + bv;
            if (c == 0)      Qo[idx] = (__bf16)(val * INV_SCALE);
            else if (c == 1) Ko[idx] = (__bf16)val;
            else             Vo[idx] = (__bf16)elu1(val);
        }
    }
}

// ---------------------------------------------------------------------------
// Kernel 2: flash attention per (b, h, 64-query tile).  WG = 128 thr (4 waves).
// Q and K tiles staged via GLOBAL_LOAD_ASYNC_TO_LDS_B128 (double-buffered K/V),
// online softmax, P -> LDS -> A-frag, P@V via WMMA.
// ---------------------------------------------------------------------------
__global__ __launch_bounds__(128) void attn_kernel(
    const __bf16* __restrict__ Qi, const __bf16* __restrict__ Ki,
    const __bf16* __restrict__ Vi, const float* __restrict__ mask,
    float* __restrict__ O)
{
    __shared__ alignas(16) __bf16 Qs [64 * 72];        // [q][hd]
    __shared__ alignas(16) __bf16 Ks [2][64 * 72];     // [kpos][hd]
    __shared__ alignas(16) __bf16 Vts[2][64 * 72];     // [hd][kpos]
    __shared__ alignas(16) __bf16 Ps [4][16 * 72];     // per-wave P

    const int tid  = threadIdx.x;
    const int lane = tid & 31;
    const int wave = tid >> 5;
    const int q0   = blockIdx.x * 64;
    const int h    = blockIdx.y;
    const int b    = blockIdx.z;

    const size_t bh = (size_t)(b * NHEADS + h) * SEQ * HEADD;
    const __bf16* Qb = Qi + bh;
    const __bf16* Kb = Ki + bh;
    const __bf16* Vb = Vi + bh;

    // ---- prologue: async Q tile + async K[0] tile, V[0] via registers ----
    v8bf rv[4];
#pragma unroll
    for (int i = 0; i < 4; ++i) {
        int q = tid + i * 128;                 // 512 x 16B chunks
        int r = q >> 3;
        int c = (q & 7) * 8;
        async_copy_b128(Qb + (size_t)(q0 + r) * HEADD + c, Qs + r * 72 + c);
        async_copy_b128(Kb + (size_t)r * HEADD + c, Ks[0] + r * 72 + c);
        rv[i] = *(const v8bf*)(Vb + (size_t)r * HEADD + c);
    }
    wait_async0();
#pragma unroll
    for (int i = 0; i < 4; ++i) {
        int q = tid + i * 128;
        int r = q >> 3;
        int c = (q & 7) * 8;
#pragma unroll
        for (int j = 0; j < 8; ++j) Vts[0][(c + j) * 72 + r] = rv[i][j];
    }
    __syncthreads();

    float rmax[8], rsum[8];
    v8f   oacc[4] = {};
#pragma unroll
    for (int r = 0; r < 8; ++r) { rmax[r] = -1e30f; rsum[r] = 0.f; }

    int cur = 0;
    for (int j0 = 0; j0 < SEQ; j0 += 64) {
        const bool has_next = (j0 + 64 < SEQ);
        v8bf rvn[4];
        if (has_next) {        // async K[next] + V[next] into regs, overlap compute
#pragma unroll
            for (int i = 0; i < 4; ++i) {
                int q = tid + i * 128;
                int r = q >> 3;
                int c = (q & 7) * 8;
                async_copy_b128(Kb + (size_t)(j0 + 64 + r) * HEADD + c,
                                Ks[cur ^ 1] + r * 72 + c);
                rvn[i] = *(const v8bf*)(Vb + (size_t)(j0 + 64 + r) * HEADD + c);
            }
        }

        // ---- scores: S = (Q/sqrt(hd)) @ K^T + mask ----
        const __bf16* Kc = Ks[cur];
        const __bf16* Vc = Vts[cur];
        v16bf a0 = load_a_frag(Qs, 72, wave * 16, 0);
        v16bf a1 = load_a_frag(Qs, 72, wave * 16, 32);
        v8f sc[4];
#pragma unroll
        for (int nb = 0; nb < 4; ++nb) {
            v8f s = {};
            s = wmma_bf16(a0, load_bt_frag(Kc, 72, nb * 16, 0),  s);
            s = wmma_bf16(a1, load_bt_frag(Kc, 72, nb * 16, 32), s);
            const float mv = mask[(size_t)b * SEQ + j0 + nb * 16 + (lane & 15)];
#pragma unroll
            for (int r = 0; r < 8; ++r) s[r] += mv;
            sc[nb] = s;
        }

        // ---- online softmax (row m = r + 8*half, reduce over 16 lanes) ----
        float alpha[8];
#pragma unroll
        for (int r = 0; r < 8; ++r) {
            float bm = fmaxf(fmaxf(sc[0][r], sc[1][r]), fmaxf(sc[2][r], sc[3][r]));
#pragma unroll
            for (int m = 1; m < 16; m <<= 1) bm = fmaxf(bm, __shfl_xor(bm, m, 32));
            const float nm = fmaxf(rmax[r], bm);
            alpha[r] = expf(rmax[r] - nm);
            rmax[r]  = nm;
        }

        float ls[8] = {0,0,0,0,0,0,0,0};
        __bf16* Pw = Ps[wave];
#pragma unroll
        for (int nb = 0; nb < 4; ++nb) {
#pragma unroll
            for (int r = 0; r < 8; ++r) {
                const float p = expf(sc[nb][r] - rmax[r]);
                ls[r] += p;
                const int m = r + ((lane >> 4) << 3);
                Pw[m * 72 + nb * 16 + (lane & 15)] = (__bf16)p;
            }
        }
#pragma unroll
        for (int r = 0; r < 8; ++r) {
            float s = ls[r];
#pragma unroll
            for (int m = 1; m < 16; m <<= 1) s += __shfl_xor(s, m, 32);
            rsum[r] = rsum[r] * alpha[r] + s;
        }
#pragma unroll
        for (int nb = 0; nb < 4; ++nb)
#pragma unroll
            for (int r = 0; r < 8; ++r) oacc[nb][r] *= alpha[r];

        __syncthreads();            // P tile visible across lanes

        // ---- O += P @ V ----
        v16bf p0 = load_a_frag(Pw, 72, 0, 0);
        v16bf p1 = load_a_frag(Pw, 72, 0, 32);
#pragma unroll
        for (int nb = 0; nb < 4; ++nb) {
            oacc[nb] = wmma_bf16(p0, load_bt_frag(Vc, 72, nb * 16, 0),  oacc[nb]);
            oacc[nb] = wmma_bf16(p1, load_bt_frag(Vc, 72, nb * 16, 32), oacc[nb]);
        }

        if (has_next) {             // finish next-buffer staging
#pragma unroll
            for (int i = 0; i < 4; ++i) {
                int q = tid + i * 128;
                int r = q >> 3;
                int c = (q & 7) * 8;
#pragma unroll
                for (int j = 0; j < 8; ++j) Vts[cur ^ 1][(c + j) * 72 + r] = rvn[i][j];
            }
            wait_async0();
            cur ^= 1;
        }
        __syncthreads();            // protect tiles + Ps for next block
    }

    // ---- normalize + write O[b][s][h*64+hd] (f32, DIM-major) ----
#pragma unroll
    for (int nb = 0; nb < 4; ++nb) {
#pragma unroll
        for (int r = 0; r < 8; ++r) {
            const int m   = r + ((lane >> 4) << 3);
            const int s   = q0 + wave * 16 + m;
            const int col = h * HEADD + nb * 16 + (lane & 15);
            O[((size_t)b * SEQ + s) * DIMN + col] = oacc[nb][r] / rsum[r];
        }
    }
}

// ---------------------------------------------------------------------------
// Kernel 3: output projection (4096x1024, K=1024), double-buffered like K1,
// writes f32 pre-layernorm values straight into d_out.
// ---------------------------------------------------------------------------
__global__ __launch_bounds__(128) void proj_gemm_kernel(
    const float* __restrict__ A, const float* __restrict__ w,
    const float* __restrict__ bias, float* __restrict__ out)
{
    __shared__ alignas(16) __bf16 As [2][64 * 40];
    __shared__ alignas(16) __bf16 Bts[2][64 * 40];

    const int tid  = threadIdx.x;
    const int lane = tid & 31;
    const int wave = tid >> 5;
    const int m0   = blockIdx.y * 64;
    const int n0   = blockIdx.x * 64;

    v8f acc[4] = {};
    float4 ra[4], rb[4];

    gemm_stage_load(A, w, DIMN, m0, n0, 0, tid, ra, rb);
    gemm_stage_store(As[0], Bts[0], tid, ra, rb);
    __syncthreads();

    int cur = 0;
    for (int k0 = 0; k0 < DIMN; k0 += 32) {
        const bool has_next = (k0 + 32 < DIMN);
        if (has_next) {
            gemm_stage_load(A, w, DIMN, m0, n0, k0 + 32, tid, ra, rb);
            __builtin_prefetch(w + (size_t)(k0 + 64) * DIMN + n0 + (lane << 2), 0, 1);
        }
        v16bf a = load_a_frag(As[cur], 40, wave * 16, 0);
#pragma unroll
        for (int nb = 0; nb < 4; ++nb)
            acc[nb] = wmma_bf16(a, load_bt_frag(Bts[cur], 40, nb * 16, 0), acc[nb]);
        if (has_next) {
            gemm_stage_store(As[cur ^ 1], Bts[cur ^ 1], tid, ra, rb);
            cur ^= 1;
        }
        __syncthreads();
    }

    const int half = (lane >> 4) << 3;
    const int ncol = lane & 15;
#pragma unroll
    for (int nb = 0; nb < 4; ++nb) {
        const int n  = n0 + nb * 16 + ncol;
        const float bv = bias[n];
#pragma unroll
        for (int r = 0; r < 8; ++r) {
            const int grow = m0 + wave * 16 + r + half;
            out[(size_t)grow * DIMN + n] = acc[nb][r] + bv;
        }
    }
}

// ---------------------------------------------------------------------------
// Kernel 4: per-row layernorm + ELU + residual, in-place on d_out.
// ---------------------------------------------------------------------------
__device__ __forceinline__ float block_sum256(float v, float* red) {
#pragma unroll
    for (int m = 1; m < 32; m <<= 1) v += __shfl_xor(v, m, 32);
    const int lane = threadIdx.x & 31;
    const int wv   = threadIdx.x >> 5;
    __syncthreads();
    if (lane == 0) red[wv] = v;
    __syncthreads();
    float t = 0.f;
#pragma unroll
    for (int i = 0; i < 8; ++i) t += red[i];
    return t;
}

__global__ __launch_bounds__(256) void ln_elu_res_kernel(
    float* __restrict__ out, const float* __restrict__ x,
    const float* __restrict__ g, const float* __restrict__ be)
{
    __shared__ float red[8];
    const int row = blockIdx.x;
    const int tid = threadIdx.x;
    float* pr = out + (size_t)row * DIMN;
    const float* px = x + (size_t)row * DIMN;

    float v[4];
    float s = 0.f;
#pragma unroll
    for (int i = 0; i < 4; ++i) { v[i] = pr[tid + i * 256]; s += v[i]; }
    s = block_sum256(s, red);
    const float mu = s * (1.0f / DIMN);

    float vs = 0.f;
#pragma unroll
    for (int i = 0; i < 4; ++i) { float d = v[i] - mu; vs += d * d; }
    vs = block_sum256(vs, red);
    const float rstd = rsqrtf(vs * (1.0f / DIMN) + LN_EPS);

#pragma unroll
    for (int i = 0; i < 4; ++i) {
        const int c = tid + i * 256;
        float y = (v[i] - mu) * rstd * g[c] + be[c];
        y = elu1(y);
        pr[c] = y + px[c];
    }
}

// ---------------------------------------------------------------------------
// Host launcher
// ---------------------------------------------------------------------------
extern "C" void kernel_launch(void* const* d_in, const int* in_sizes, int n_in,
                              void* d_out, int out_size, void* d_ws, size_t ws_size,
                              hipStream_t stream) {
    const float* x     = (const float*)d_in[0];
    const float* mask  = (const float*)d_in[1];
    const float* w_qkv = (const float*)d_in[2];
    const float* b_qkv = (const float*)d_in[3];
    const float* w_out = (const float*)d_in[4];
    const float* b_out = (const float*)d_in[5];
    const float* ln_g  = (const float*)d_in[6];
    const float* ln_b  = (const float*)d_in[7];
    float* out = (float*)d_out;

    // workspace: Q|K|V bf16 head-major (8 MB each) + attention O f32 (16 MB)
    const size_t elems = (size_t)BATCH * NHEADS * SEQ * HEADD;
    __bf16* Q = (__bf16*)d_ws;
    __bf16* K = Q + elems;
    __bf16* V = K + elems;
    float*  O = (float*)(V + elems);

    {
        dim3 grid(QKVN / 64, MROWS / 64);
        qkv_gemm_kernel<<<grid, 128, 0, stream>>>(x, w_qkv, b_qkv, Q, K, V);
    }
    {
        dim3 grid(SEQ / 64, NHEADS, BATCH);
        attn_kernel<<<grid, 128, 0, stream>>>(Q, K, V, mask, O);
    }
    {
        dim3 grid(DIMN / 64, MROWS / 64);
        proj_gemm_kernel<<<grid, 128, 0, stream>>>(O, w_out, b_out, out);
    }
    {
        ln_elu_res_kernel<<<MROWS, 256, 0, stream>>>(out, x, ln_g, ln_b);
    }
}
